// PainnMessage_76879914598801
// MI455X (gfx1250) — compile-verified, matrix operand order checked by reading, not run on hardware
//
#include <hip/hip_runtime.h>

// ---------------------------------------------------------------------------
// PaiNN message layer for MI455X (gfx1250, wave32, WMMA).
//  - Weights pre-packed once per launch into WMMA B-fragment bf16 layout so
//    the GEMM inner loop is pure {2x ds_load_b128 (A) + 2x global_load_b128
//    (B) + v_wmma_f32_16x16x32_bf16} with no conversion VALU.
//  - Fused 2-layer MLP: layer-1 activations live in LDS only.
//  - Edge phase: coalesced gather/gate + hardware f32 atomic scatter-add
//    (bandwidth roofline: ~4.3 GB @ 23.3 TB/s).
// ---------------------------------------------------------------------------

typedef __attribute__((ext_vector_type(16))) __bf16 v16bf;
typedef __attribute__((ext_vector_type(8)))  __bf16 v8bf;
typedef __attribute__((ext_vector_type(4)))  __bf16 v4bf;
typedef __attribute__((ext_vector_type(8)))  float  v8f;

#define HID   128
#define OUT3H 384

__device__ __forceinline__ float prelu(float x, float a) {
    return x > 0.f ? x : a * x;
}

__device__ __forceinline__ void atomAddF(float* p, float v) {
    // non-returning relaxed agent-scope add -> global_atomic_add_f32 (no RTN)
    __hip_atomic_fetch_add(p, v, __ATOMIC_RELAXED, __HIP_MEMORY_SCOPE_AGENT);
}

// WMMA 16-bit fragment index mapping (ISA 7.12.2): element i of the v16bf,
// given the lane's K-half offset kh, sits at K = ((i>>1)&4 ? 16:0) + kh +
// (((i>>1)&3)<<1) + (i&1).
__device__ __host__ __forceinline__ int frag_k(int i, int kh) {
    int v = i >> 1;
    return ((v & 4) ? 16 : 0) + kh + ((v & 3) << 1) + (i & 1);
}

// ---------------------------------------------------------------------------
// Pack W[K,Nn] (f32, row-major) into B-fragment bf16 layout:
//   out[((nt*KS + ks)*32 + lane)*16 + i] = W[ks*32 + frag_k(i,kh)][nt*16 + n16]
// so a wave's B tile is one contiguous, coalesced 1 KB block.
// ---------------------------------------------------------------------------
__global__ void pack_weights_kernel(const float* __restrict__ W, int K, int Nn,
                                    int KS, __bf16* __restrict__ out)
{
    const int total = (Nn / 16) * KS * 512;
    for (int t = blockIdx.x * blockDim.x + threadIdx.x; t < total;
         t += gridDim.x * blockDim.x) {
        int i    = t & 15;
        int lane = (t >> 4) & 31;
        int blk  = t >> 9;          // nt*KS + ks
        int ks   = blk % KS;
        int nt   = blk / KS;
        int n    = nt * 16 + (lane & 15);
        int kh   = (lane >> 4) * 8;
        int k    = ks * 32 + frag_k(i, kh);
        float v  = (k < K) ? W[(size_t)k * Nn + n] : 0.f;
        out[t] = (__bf16)v;
    }
}

// ---------------------------------------------------------------------------
// Fused  out = prelu( prelu(X @ W1 + b1, a1) @ W2 + b2, a2 )
//   X: [M,K1] f32, W1p/W2p: fragment-packed bf16, out: [M,384] f32
// One 256-thread block (8 wave32) computes 64 rows.
// ---------------------------------------------------------------------------
template<int K1PAD>
__global__ __launch_bounds__(256)
void mlp2_wmma_kernel(const float* __restrict__ X, int M, int K1,
                      const __bf16* __restrict__ W1p, const float* __restrict__ b1,
                      const float* __restrict__ a1p,
                      const __bf16* __restrict__ W2p, const float* __restrict__ b2,
                      const float* __restrict__ a2p,
                      float* __restrict__ out)
{
    constexpr int KS1 = K1PAD / 32;
    __shared__ __bf16 Xs[64][K1PAD];   // input tile (bf16)
    __shared__ __bf16 Hs[64][HID];     // layer-1 activations (bf16)

    const int tid  = threadIdx.x;
    const int wid  = tid >> 5;         // wave id 0..7 (wave32)
    const int lane = tid & 31;
    const int mloc = lane & 15;        // M within 16x16 tile
    const int half = lane >> 4;
    const int kh   = half * 8;
    const int row0 = blockIdx.x * 64;
    const float a1 = a1p[0], a2 = a2p[0];

    // warm L2/L0 with this wave's stage-2 packed weights (global_prefetch_b8)
    #pragma unroll
    for (int c = 0; c < 3; ++c)
        __builtin_prefetch((const char*)(W2p + (size_t)((wid + c * 8) * 4) * 512)
                               + lane * 64, 0, 0);

    // cooperative X load: float4 -> 4x bf16 -> single 8-byte LDS store
    for (int t = tid; t < 64 * (K1PAD / 4); t += 256) {
        int r  = t / (K1PAD / 4);
        int c  = (t % (K1PAD / 4)) * 4;
        int row = row0 + r;
        float4 v = make_float4(0.f, 0.f, 0.f, 0.f);
        if (row < M && c + 3 < K1) {
            v = *(const float4*)(X + (size_t)row * K1 + c);
        } else if (row < M) {
            float tmp[4] = {0.f, 0.f, 0.f, 0.f};
            for (int j = 0; j < 4; ++j)
                if (c + j < K1) tmp[j] = X[(size_t)row * K1 + c + j];
            v = make_float4(tmp[0], tmp[1], tmp[2], tmp[3]);
        }
        v4bf p;
        p[0] = (__bf16)v.x; p[1] = (__bf16)v.y;
        p[2] = (__bf16)v.z; p[3] = (__bf16)v.w;
        *(v4bf*)&Xs[r][c] = p;
    }
    __syncthreads();

    const v16bf* W1v = (const v16bf*)W1p;
    const v16bf* W2v = (const v16bf*)W2p;

    // ---- Stage 1: Hs = prelu(X @ W1 + b1) ; wave owns cols [wid*16,+16) ----
    {
        const int   n    = wid * 16 + mloc;
        const float bias = b1[n];
        for (int rt = 0; rt < 4; ++rt) {
            v8f acc = {};
            #pragma unroll
            for (int ks = 0; ks < KS1; ++ks) {
                const __bf16* arow = &Xs[rt * 16 + mloc][ks * 32];
                v8bf lo = *(const v8bf*)(arow + kh);        // ds_load_b128
                v8bf hi = *(const v8bf*)(arow + 16 + kh);   // ds_load_b128
                v16bf a = __builtin_shufflevector(lo, hi, 0, 1, 2, 3, 4, 5, 6, 7,
                                                  8, 9, 10, 11, 12, 13, 14, 15);
                v16bf b = W1v[(wid * KS1 + ks) * 32 + lane]; // 2x global_load_b128
                acc = __builtin_amdgcn_wmma_f32_16x16x32_bf16(
                          false, a, false, b, (short)0, acc, false, false);
            }
            #pragma unroll
            for (int r = 0; r < 8; ++r) {
                int m = rt * 16 + half * 8 + r;              // C layout
                Hs[m][n] = (__bf16)prelu(acc[r] + bias, a1);
            }
        }
    }
    __syncthreads();

    // ---- Stage 2: out = prelu(Hs @ W2 + b2) ; 24 col tiles, 3 per wave ----
    for (int c = 0; c < 3; ++c) {
        const int   nt   = wid + c * 8;
        const int   n    = nt * 16 + mloc;
        const float bias = b2[n];
        for (int rt = 0; rt < 4; ++rt) {
            v8f acc = {};
            #pragma unroll
            for (int ks = 0; ks < 4; ++ks) {
                const __bf16* arow = &Hs[rt * 16 + mloc][ks * 32];
                v8bf lo = *(const v8bf*)(arow + kh);
                v8bf hi = *(const v8bf*)(arow + 16 + kh);
                v16bf a = __builtin_shufflevector(lo, hi, 0, 1, 2, 3, 4, 5, 6, 7,
                                                  8, 9, 10, 11, 12, 13, 14, 15);
                v16bf b = W2v[(nt * 4 + ks) * 32 + lane];
                acc = __builtin_amdgcn_wmma_f32_16x16x32_bf16(
                          false, a, false, b, (short)0, acc, false, false);
            }
            #pragma unroll
            for (int r = 0; r < 8; ++r) {
                int row = row0 + rt * 16 + half * 8 + r;
                if (row < M)
                    out[(size_t)row * OUT3H + n] = prelu(acc[r] + bias, a2);
            }
        }
    }
}

// ---------------------------------------------------------------------------
// out init:  d_out = [node_scalar | node_vector]  (residual term)
// ---------------------------------------------------------------------------
__global__ void init_out_kernel(const float* __restrict__ ns,
                                const float* __restrict__ nv,
                                float* __restrict__ out,
                                int nScalar, int nTotal)
{
    for (int i = blockIdx.x * blockDim.x + threadIdx.x; i < nTotal;
         i += gridDim.x * blockDim.x)
        out[i] = (i < nScalar) ? ns[i] : nv[i - nScalar];
}

// ---------------------------------------------------------------------------
// Per-edge gather + gate + scatter-add. One block per edge, h = threadIdx.x.
// ---------------------------------------------------------------------------
__global__ __launch_bounds__(128)
void edge_scatter_kernel(const int*   __restrict__ edge,   // [E,2]
                         const float* __restrict__ diff,   // [E,3]
                         const float* __restrict__ dist,   // [E]
                         const float* __restrict__ scal,   // [N,384] scalar_out
                         const float* __restrict__ filt,   // [E,384] filter_w
                         const float* __restrict__ nvec,   // [N,3,128]
                         float* __restrict__ outS,         // [N,128]
                         float* __restrict__ outV,         // [N,3,128]
                         int E)
{
    const int e = blockIdx.x;
    if (e >= E) return;
    const int h   = threadIdx.x;
    const int dst = edge[2 * e + 0];
    const int src = edge[2 * e + 1];
    const size_t fb = (size_t)e   * OUT3H;
    const size_t sb = (size_t)src * OUT3H;

    const float gv = scal[sb +       h] * filt[fb +       h];  // gate_vector
    const float ms = scal[sb + 128 + h] * filt[fb + 128 + h];  // message_scalar
    const float ge = scal[sb + 256 + h] * filt[fb + 256 + h];  // gate_edge
    const float inv = 1.f / dist[e];

    atomAddF(&outS[(size_t)dst * HID + h], ms);
    #pragma unroll
    for (int d = 0; d < 3; ++d) {
        float u  = diff[3 * e + d] * inv;
        float mv = nvec[((size_t)src * 3 + d) * HID + h] * gv + u * ge;
        atomAddF(&outV[((size_t)dst * 3 + d) * HID + h], mv);
    }
}

// ---------------------------------------------------------------------------
extern "C" void kernel_launch(void* const* d_in, const int* in_sizes, int n_in,
                              void* d_out, int out_size, void* d_ws, size_t ws_size,
                              hipStream_t stream)
{
    const int N = 50000, H = 128, RBF = 20, E_CHE = 400000, E_VDW = 200000;

    const float* node_scalar = (const float*)d_in[0];
    const float* node_vector = (const float*)d_in[1];
    const int*   che_edge    = (const int*)  d_in[2];
    const float* che_diff    = (const float*)d_in[3];
    const float* che_dist    = (const float*)d_in[4];
    const float* che_rbf     = (const float*)d_in[5];
    const int*   vdw_edge    = (const int*)  d_in[6];
    const float* vdw_diff    = (const float*)d_in[7];
    const float* vdw_dist    = (const float*)d_in[8];
    const float* vdw_rbf     = (const float*)d_in[9];
    // P[0..5]=che_s, P[6..11]=che_f, P[12..17]=vdw_s, P[18..23]=vdw_f
    const float* P[24];
    for (int i = 0; i < 24; ++i) P[i] = (const float*)d_in[10 + i];

    float* out  = (float*)d_out;
    float* outS = out;                            // [N,128]
    float* outV = out + (size_t)N * H;            // [N,3,128]

    // --- workspace layout ---
    // packed bf16 weights first (<=512 KB), then scalar_out, then filter buf
    __bf16* wp    = (__bf16*)d_ws;
    const size_t SZ_W1N = 4 * 8 * 512;            // K=128 pad128 -> Nn=128
    const size_t SZ_W2  = 4 * 24 * 512;           // K=128 -> Nn=384
    const size_t SZ_W1F = 1 * 8 * 512;            // K=20 pad32 -> Nn=128
    __bf16* cheS_W1 = wp;                 __bf16* cheS_W2 = cheS_W1 + SZ_W1N;
    __bf16* cheF_W1 = cheS_W2 + SZ_W2;    __bf16* cheF_W2 = cheF_W1 + SZ_W1F;
    __bf16* vdwS_W1 = cheF_W2 + SZ_W2;    __bf16* vdwS_W2 = vdwS_W1 + SZ_W1N;
    __bf16* vdwF_W1 = vdwS_W2 + SZ_W2;    __bf16* vdwF_W2 = vdwF_W1 + SZ_W1F;
    float* ws_scal = (float*)((char*)d_ws + 512 * 1024);   // [N,384]
    float* ws_filt = ws_scal + (size_t)N * OUT3H;          // [E_CHE,384]

    // residual init: out = (node_scalar, node_vector)
    init_out_kernel<<<2048, 256, 0, stream>>>(node_scalar, node_vector, out,
                                              N * H, N * H + N * 3 * H);

    // --- pack all 8 weight matrices to WMMA B-fragment bf16 layout ---
    pack_weights_kernel<<<64, 256, 0, stream>>>(P[0],  H,   H,     4, cheS_W1);
    pack_weights_kernel<<<96, 256, 0, stream>>>(P[3],  H,   OUT3H, 4, cheS_W2);
    pack_weights_kernel<<<16, 256, 0, stream>>>(P[6],  RBF, H,     1, cheF_W1);
    pack_weights_kernel<<<96, 256, 0, stream>>>(P[9],  H,   OUT3H, 4, cheF_W2);
    pack_weights_kernel<<<64, 256, 0, stream>>>(P[12], H,   H,     4, vdwS_W1);
    pack_weights_kernel<<<96, 256, 0, stream>>>(P[15], H,   OUT3H, 4, vdwS_W2);
    pack_weights_kernel<<<16, 256, 0, stream>>>(P[18], RBF, H,     1, vdwF_W1);
    pack_weights_kernel<<<96, 256, 0, stream>>>(P[21], H,   OUT3H, 4, vdwF_W2);

    // ---- chemical branch ----
    mlp2_wmma_kernel<128><<<(N + 63) / 64, 256, 0, stream>>>(
        node_scalar, N, H, cheS_W1, P[1], P[2], cheS_W2, P[4], P[5], ws_scal);
    mlp2_wmma_kernel<32><<<(E_CHE + 63) / 64, 256, 0, stream>>>(
        che_rbf, E_CHE, RBF, cheF_W1, P[7], P[8], cheF_W2, P[10], P[11], ws_filt);
    edge_scatter_kernel<<<E_CHE, 128, 0, stream>>>(
        che_edge, che_diff, che_dist, ws_scal, ws_filt, node_vector,
        outS, outV, E_CHE);

    // ---- van-der-Waals branch (workspace reused, stream-ordered) ----
    mlp2_wmma_kernel<128><<<(N + 63) / 64, 256, 0, stream>>>(
        node_scalar, N, H, vdwS_W1, P[13], P[14], vdwS_W2, P[16], P[17], ws_scal);
    mlp2_wmma_kernel<32><<<(E_VDW + 63) / 64, 256, 0, stream>>>(
        vdw_rbf, E_VDW, RBF, vdwF_W1, P[19], P[20], vdwF_W2, P[22], P[23], ws_filt);
    edge_scatter_kernel<<<E_VDW, 128, 0, stream>>>(
        vdw_edge, vdw_diff, vdw_dist, ws_scal, ws_filt, node_vector,
        outS, outV, E_VDW);
}
